// Convolution_30408368455936
// MI455X (gfx1250) — matrix operand order repaired and checked by reading, not compiled, not run on hardware
//
#include <hip/hip_runtime.h>
#include <hip/hip_bf16.h>

typedef __attribute__((ext_vector_type(16))) _Float16 v16h;
typedef __attribute__((ext_vector_type(8)))  float    v8f;
typedef __attribute__((ext_vector_type(4)))  unsigned int u32x4;
typedef __attribute__((ext_vector_type(4)))  float    f32x4;

// ---------------- geometry constants ----------------
// x padded: [2][52][52][52][64] f16
#define SZP 64
#define SYP (52*64)
#define SXP (52*52*64)
#define SNP (52*52*52*64)           // 8,998,912 halves
#define PAD_BYTES (2*52*52*52*64*2) // 35,995,648 bytes
// output logical: [2][48][48][48][64] f32; M = 2*48^3 = 221184 positions

union FragU { u32x4 q[2]; v16h v; };

__device__ inline v16h loadA(const _Float16* p) {
  // A fragment (16-bit, 16x32): elems 0..7 = K(+8*hi)+0..7, elems 8..15 = K+16(+8*hi)+0..7
  FragU f;
  f.q[0] = *(const u32x4*)(p);
  f.q[1] = *(const u32x4*)(p + 16);
  return f.v;
}

__device__ inline v16h loadB(const _Float16* p) {
  // B fragment: lane holds one column, elems 0..7 = K(+8*hi)+0..7, elems 8..15 = K+16(+8*hi)+0..7
  FragU f;
  f.q[0] = *(const u32x4*)(p);
  f.q[1] = *(const u32x4*)(p + 16);
  return f.v;
}

// ---------------- pre-pass 1: zero-pad + cast to f16 ----------------
__global__ void __launch_bounds__(256) pad_cast(const float* __restrict__ x,
                                                _Float16* __restrict__ xp) {
  int t = blockIdx.x * 256 + threadIdx.x;     // one thread = 8 halves (16B)
  int f = t * 8;
  int v = f >> 6;                             // voxel index in padded volume
  int c = f & 63;                             // channel base (multiple of 8)
  int pz = v % 52; int v2 = v / 52;
  int py = v2 % 52; int v3 = v2 / 52;
  int px = v3 % 52; int n  = v3 / 52;
  union { _Float16 h[8]; u32x4 q; } o;
  if (px >= 2 && px < 50 && py >= 2 && py < 50 && pz >= 2 && pz < 50) {
    const float* s = x + ((((long)n*48 + (px-2))*48 + (py-2))*48 + (pz-2))*64 + c;
    f32x4 a = *(const f32x4*)(s);
    f32x4 b = *(const f32x4*)(s + 4);
    o.h[0]=(_Float16)a.x; o.h[1]=(_Float16)a.y; o.h[2]=(_Float16)a.z; o.h[3]=(_Float16)a.w;
    o.h[4]=(_Float16)b.x; o.h[5]=(_Float16)b.y; o.h[6]=(_Float16)b.z; o.h[7]=(_Float16)b.w;
  } else {
    o.q = (u32x4){0u,0u,0u,0u};
  }
  *(u32x4*)(xp + f) = o.q;
}

// ---------------- pre-pass 2: build kernel, store transposed f16 kT[tap][o][i] ----------------
__global__ void __launch_bounds__(256) build_k(const float* __restrict__ w000,
                                               const float* __restrict__ w011,
                                               const float* __restrict__ w101,
                                               const float* __restrict__ w110,
                                               const float* __restrict__ sc0,
                                               const float* __restrict__ sc1,
                                               _Float16* __restrict__ kT) {
  int tap = blockIdx.x;                 // 0..124
  int dx = tap / 25, dy = (tap / 5) % 5, dz = tap % 5;
  float cx = (float)(dx - 2), cy = (float)(dy - 2), cz = (float)(dz - 2);
  float norm = sqrtf(cx*cx + cy*cy + cz*cz);
  float sh[3];
  if (norm > 0.0f) {
    float inv = 1.7320508075688772f / norm;   // sqrt(3)*nvec
    sh[0] = cx*inv; sh[1] = cy*inv; sh[2] = cz*inv;
  } else { sh[0] = sh[1] = sh[2] = 0.0f; }
  float emb[4];
  #pragma unroll
  for (int b = 0; b < 4; ++b) {
    float vb = 0.5f * (float)(b + 1);
    float d  = (norm - vb) * 2.0f;             // /step, step = 0.5
    float dd = fminf(fmaxf(d, -1.0f + 1e-9f), 1.0f - 1e-9f);
    emb[b] = (fabsf(d) < 1.0f)
           ? 1.14136f * 7.38905609893065f * expf(-1.0f / (1.0f - dd*dd))
           : 0.0f;
  }
  const bool  center   = (tap == 62);
  const float inv_nlat = 1.0f / 125.0f;
  const float inv_s3   = 0.5773502691896258f;

  for (int idx = threadIdx.x; idx < 4096; idx += 256) {
    int i = idx >> 6, o = idx & 63;            // i = input row, o = output col
    float val;
    if (!center) {
      if (i < 16 && o < 16) {                  // 0e x 0e -> 0e
        float r = 0.f;
        #pragma unroll
        for (int b = 0; b < 4; ++b) r += emb[b] * w000[(b*16 + i)*16 + o];
        val = r * inv_nlat;
      } else if (i < 16) {                     // 0e x 1o -> 1o
        int oc = o - 16, w = oc / 3, kc = oc % 3;
        float r = 0.f;
        #pragma unroll
        for (int b = 0; b < 4; ++b) r += emb[b] * w011[(b*16 + i)*16 + w];
        val = r * inv_nlat * sh[kc];
      } else if (o < 16) {                     // 1o x 1o -> 0e
        int ic = i - 16, u = ic / 3, icp = ic % 3;
        float r = 0.f;
        #pragma unroll
        for (int b = 0; b < 4; ++b) r += emb[b] * w110[(b*16 + u)*16 + o];
        val = r * inv_nlat * sh[icp] * inv_s3;
      } else {                                 // 1o x 0e -> 1o (delta on components)
        int ic = i - 16, u = ic / 3, icp = ic % 3;
        int oc = o - 16, w = oc / 3, kc  = oc % 3;
        if (icp == kc) {
          float r = 0.f;
          #pragma unroll
          for (int b = 0; b < 4; ++b) r += emb[b] * w101[(b*16 + u)*16 + w];
          val = r * inv_nlat;
        } else val = 0.f;
      }
    } else {                                   // self-connection linear, std 1/sqrt(16)
      if (i < 16 && o < 16) {
        val = sc0[i*16 + o] * 0.25f;
      } else if (i >= 16 && o >= 16) {
        int ic = i - 16, u = ic / 3, icp = ic % 3;
        int oc = o - 16, w = oc / 3, kc  = oc % 3;
        val = (icp == kc) ? sc1[u*16 + w] * 0.25f : 0.f;
      } else val = 0.f;
    }
    kT[tap*4096 + o*64 + i] = (_Float16)val;   // transposed: i contiguous
  }
}

// ---------------- main conv: implicit GEMM with v_wmma_f32_16x16x32_f16 ----------------
// Each wave: M = 64 output positions (four 16-row tiles) x N = 64 outputs.
// Per tap: 16 A-loads + 16 B-loads (b128) feeding 32 WMMAs -> 1.0 WMMA/VMEM issue ratio.
__global__ void __launch_bounds__(256) conv_wmma(const _Float16* __restrict__ xp,
                                                 const _Float16* __restrict__ kT,
                                                 float* __restrict__ out) {
  const int lane = threadIdx.x & 31;
  const int wave = threadIdx.x >> 5;
  const int hi   = lane >> 4;     // 0/1: half-wave selector
  const int lo   = lane & 15;

  const int m0 = (blockIdx.x * 8 + wave) * 64;   // 64 output positions per wave

  const _Float16* baseA[4];
  int mrow[4];
  #pragma unroll
  for (int t = 0; t < 4; ++t) {
    int m = m0 + t*16;                      // 16-run never crosses a z-line (48 % 16 == 0)
    int n   = m / 110592; int rem = m % 110592;
    int xi  = rem / 2304; int rem2 = rem % 2304;
    int yi  = rem2 / 48;  int z0   = rem2 % 48;
    // padded coord of this lane's row (tap offset added later; pad=2 cancels dx-2)
    baseA[t] = xp + n*SNP + xi*SXP + yi*SYP + (z0 + lo)*SZP;
    mrow[t] = m;
  }
  // B fragment lane base: column = lo (within n-tile), K sub-offset 8*hi
  const _Float16* bLane = kT + lo*64 + hi*8;
  const int aOff = hi * 8;                  // A channel sub-offset within K-chunk

  v8f acc[4][4];
  #pragma unroll
  for (int t = 0; t < 4; ++t)
    #pragma unroll
    for (int nt = 0; nt < 4; ++nt)
      acc[t][nt] = (v8f){0.f,0.f,0.f,0.f,0.f,0.f,0.f,0.f};

  for (int dx = 0; dx < 5; ++dx)
    for (int dy = 0; dy < 5; ++dy)
      for (int dz = 0; dz < 5; ++dz) {
        const int off = dx*SXP + dy*SYP + dz*SZP;
        const _Float16* pB = bLane + ((dx*5 + dy)*5 + dz)*4096;
        #pragma unroll
        for (int kt = 0; kt < 2; ++kt) {          // K = 64 channels, two 32-chunks
          v16h a0 = loadA(baseA[0] + off + kt*32 + aOff);
          v16h a1 = loadA(baseA[1] + off + kt*32 + aOff);
          v16h a2 = loadA(baseA[2] + off + kt*32 + aOff);
          v16h a3 = loadA(baseA[3] + off + kt*32 + aOff);
          #pragma unroll
          for (int nt = 0; nt < 4; ++nt) {        // N = 64 outputs, four 16-tiles
            v16h b = loadB(pB + nt*1024 + kt*32);
            acc[0][nt] = __builtin_amdgcn_wmma_f32_16x16x32_f16(
                false, a0, false, b, (short)0, acc[0][nt], false, false);
            acc[1][nt] = __builtin_amdgcn_wmma_f32_16x16x32_f16(
                false, a1, false, b, (short)0, acc[1][nt], false, false);
            acc[2][nt] = __builtin_amdgcn_wmma_f32_16x16x32_f16(
                false, a2, false, b, (short)0, acc[2][nt], false, false);
            acc[3][nt] = __builtin_amdgcn_wmma_f32_16x16x32_f16(
                false, a3, false, b, (short)0, acc[3][nt], false, false);
          }
        }
      }

  // store: C/D layout — VGPR j: M = j + 8*hi, N = lo
  #pragma unroll
  for (int t = 0; t < 4; ++t) {
    int mb = mrow[t] + 8*hi;
    #pragma unroll
    for (int nt = 0; nt < 4; ++nt)
      #pragma unroll
      for (int j = 0; j < 8; ++j)
        out[(mb + j)*64 + nt*16 + lo] = acc[t][nt][j];
  }
}

extern "C" void kernel_launch(void* const* d_in, const int* in_sizes, int n_in,
                              void* d_out, int out_size, void* d_ws, size_t ws_size,
                              hipStream_t stream) {
  const float* x    = (const float*)d_in[0];
  const float* w000 = (const float*)d_in[1];
  const float* w011 = (const float*)d_in[2];
  const float* w101 = (const float*)d_in[3];
  const float* w110 = (const float*)d_in[4];
  const float* sc0  = (const float*)d_in[5];
  const float* sc1  = (const float*)d_in[6];
  float* out = (float*)d_out;

  _Float16* xp = (_Float16*)d_ws;                              // 35,995,648 B
  _Float16* kT = (_Float16*)((char*)d_ws + PAD_BYTES);         // 1,024,000 B

  // 1) pad + f16 cast: 2*52^3*64 halves, 8 halves/thread -> 2,249,728 threads
  pad_cast<<<8788, 256, 0, stream>>>(x, xp);
  // 2) build 125 x 64 x 64 kernel (transposed, f16)
  build_k<<<125, 256, 0, stream>>>(w000, w011, w101, w110, sc0, sc1, kT);
  // 3) implicit-GEMM conv: 221184 positions / (8 waves * 64) = 432 blocks
  conv_wmma<<<432, 256, 0, stream>>>(xp, kT, out);
}
